// CONAD_25907242729574
// MI455X (gfx1250) — matrix operand order repaired and checked by reading, not compiled, not run on hardware
//
#include <hip/hip_runtime.h>
#include <hip/hip_bf16.h>

// ---------------------------------------------------------------------------
// Problem constants (match reference)
// ---------------------------------------------------------------------------
#define NN    8192
#define FF    1433
#define HH    2
#define HID   128
#define OUTD  64
#define SLOPE_ATTN 0.2f
#define SLOPE_ACT  0.01f

typedef __attribute__((ext_vector_type(16))) __bf16 v16bf;
typedef __attribute__((ext_vector_type(8)))  float  v8f;

__device__ __forceinline__ float lrelu(float x, float s) { return x > 0.f ? x : s * x; }

// Monotone float -> uint encoding for atomicMax-based segment_max.
__device__ __forceinline__ unsigned enc_f(float f) {
    unsigned u = __float_as_uint(f);
    return (u >> 31) ? ~u : (u | 0x80000000u);
}
__device__ __forceinline__ float dec_f(unsigned k) {
    unsigned u = (k & 0x80000000u) ? (k ^ 0x80000000u) : ~k;
    return __uint_as_float(u);
}

__device__ __forceinline__ int imin(int a, int b) { return a < b ? a : b; }

// ---------------------------------------------------------------------------
// bf16 WMMA GEMM: C[M,NnC] = A[M,K] * op(B), fp32 in memory, bf16 math, f32 acc.
//   NnC: COMPILE-TIME output width -> all fragment loads/stores use immediate
//        offsets off stable base pointers (no per-element 64-bit address math).
//   BT=false: B is [K,NnC] row-major.   BT=true: B is [NnC,K] row-major (B^T).
//   NT=true : nontemporal stores (268MB a_hat stream must not thrash L2).
//
// One 16x32 C slab per wave (two 16x16 tiles sharing the A fragment).
// Fragment lane layouts used here (A/B in contiguous k order, same for both):
//   A 16x32 bf16: lane row = lane%16; elem e -> K = e + (lane>=16?16:0)
//   B 32x16 bf16: lane col = lane%16; elem e -> K = e + (lane>=16?16:0)
//   C f32 16x16 : lane col = lane%16; vgpr r -> row = r + (lane>=16?8:0)
// Out-of-range rows/cols are CLAMPED (they only pollute un-stored outputs);
// only the K tail needs zero-masking, done branch-free with clamped indices.
// ---------------------------------------------------------------------------
template <int NnC, bool BT, bool NT>
__global__ void gemm_wmma(const float* __restrict__ A, const float* __restrict__ B,
                          float* __restrict__ C, int M, int K) {
    const int wave = blockIdx.x * (blockDim.x >> 5) + (threadIdx.x >> 5);
    const int lane = threadIdx.x & 31;
    constexpr int tiles_n2 = (NnC + 31) >> 5;        // 32-wide slabs
    const int tiles_m = (M + 15) >> 4;
    if (wave >= tiles_m * tiles_n2) return;          // wave-uniform: EXEC stays full
    const int tm = wave / tiles_n2, tn2 = wave % tiles_n2;
    const int l16  = lane & 15;
    const int hi8  = (lane >> 4) * 8;
    const int hi16 = (lane >> 4) * 16;

    const int row  = imin(tm * 16 + l16, M - 1);     // clamped A row for this lane
    const int col0 = tn2 * 32 + l16;                 // B cols for the two tiles
    const int col1 = col0 + 16;
    const int c0c  = imin(col0, NnC - 1);
    const int c1c  = imin(col1, NnC - 1);

    const float* Ar = A + (size_t)row * K;           // for the K tail
    const float* Ab = Ar + hi16;                     // steady-state base (imm offsets)
    const float *B0, *B1;
    if (BT) { B0 = B + (size_t)c0c * K + hi16;  B1 = B + (size_t)c1c * K + hi16; }
    else    { B0 = B + (size_t)hi16 * NnC + c0c; B1 = B + (size_t)hi16 * NnC + c1c; }

    v8f acc0 = {}, acc1 = {};
    const int kfull = K & ~31;

    for (int k0 = 0; k0 < kfull; k0 += 32) {
        v16bf a, b0, b1;
#pragma unroll
        for (int e = 0; e < 8; ++e) {                // imm offsets: e, 8+e
            a[e]     = (__bf16)Ab[e];
            a[e + 8] = (__bf16)Ab[8 + e];
        }
#pragma unroll
        for (int e = 0; e < 16; ++e) {               // imm offsets: e or e*NnC
            if (BT) {
                b0[e] = (__bf16)B0[e];
                b1[e] = (__bf16)B1[e];
            } else {
                b0[e] = (__bf16)B0[(size_t)e * NnC];
                b1[e] = (__bf16)B1[(size_t)e * NnC];
            }
        }
        acc0 = __builtin_amdgcn_wmma_f32_16x16x32_bf16(false, a, false, b0,
                                                       (short)0, acc0, false, false);
        acc1 = __builtin_amdgcn_wmma_f32_16x16x32_bf16(false, a, false, b1,
                                                       (short)0, acc1, false, false);
        Ab += 32;
        B0 += BT ? 32 : 32 * (size_t)NnC;
        B1 += BT ? 32 : 32 * (size_t)NnC;
    }

    if (kfull < K) {                                 // K tail: clamped idx + selects
        v16bf a, b0, b1;
#pragma unroll
        for (int e = 0; e < 16; ++e) {
            const int ka = kfull + e + hi16;
            const float av = Ar[imin(ka, K - 1)];
            a[e] = (__bf16)(ka < K ? av : 0.f);
            const int kb  = kfull + hi16 + e;
            const int kbc = imin(kb, K - 1);
            float bv0, bv1;
            if (BT) {
                bv0 = B[(size_t)c0c * K + kbc];
                bv1 = B[(size_t)c1c * K + kbc];
            } else {
                bv0 = B[(size_t)kbc * NnC + c0c];
                bv1 = B[(size_t)kbc * NnC + c1c];
            }
            b0[e] = (__bf16)(kb < K ? bv0 : 0.f);
            b1[e] = (__bf16)(kb < K ? bv1 : 0.f);
        }
        acc0 = __builtin_amdgcn_wmma_f32_16x16x32_bf16(false, a, false, b0,
                                                       (short)0, acc0, false, false);
        acc1 = __builtin_amdgcn_wmma_f32_16x16x32_bf16(false, a, false, b1,
                                                       (short)0, acc1, false, false);
    }

    // Store: base pointer per lane, row offsets r*NnC are immediates.
    float* Cp = C + (size_t)(tm * 16 + hi8) * NnC;
#pragma unroll
    for (int r = 0; r < 8; ++r) {
        const int rr = tm * 16 + r + hi8;
        if (rr < M) {
            float* p = Cp + (size_t)r * NnC;
            if (NnC % 32 == 0 || col0 < NnC) {
                if (NT) __builtin_nontemporal_store(acc0[r], p + col0);
                else p[col0] = acc0[r];
            }
            if (NnC % 32 == 0 || col1 < NnC) {
                if (NT) __builtin_nontemporal_store(acc1[r], p + col1);
                else p[col1] = acc1[r];
            }
        }
    }
}

// ---------------------------------------------------------------------------
// el[n,h] = sum_d h[n,h,d]*al[h,d] ; er likewise.  One wave32 per (node,head).
// ---------------------------------------------------------------------------
__global__ void attn_dots(const float* __restrict__ h, const float* __restrict__ al,
                          const float* __restrict__ ar, float* __restrict__ el,
                          float* __restrict__ er, int n_nodes, int nh, int D) {
    int wave = blockIdx.x * (blockDim.x >> 5) + (threadIdx.x >> 5);
    int lane = threadIdx.x & 31;
    if (wave >= n_nodes * nh) return;
    int n = wave / nh, hd = wave % nh;
    const float* hp  = h  + ((size_t)n * nh + hd) * D;
    const float* alp = al + (size_t)hd * D;
    const float* arp = ar + (size_t)hd * D;
    float sl = 0.f, sr = 0.f;
    for (int d = lane; d < D; d += 32) {
        float v = hp[d];
        sl += v * alp[d];
        sr += v * arp[d];
    }
#pragma unroll
    for (int o = 16; o > 0; o >>= 1) {
        sl += __shfl_xor(sl, o, 32);
        sr += __shfl_xor(sr, o, 32);
    }
    if (lane == 0) { el[wave] = sl; er[wave] = sr; }
}

// segment_max over dst via atomicMax on monotone-encoded floats
__global__ void edge_max_k(const float* __restrict__ el, const float* __restrict__ er,
                           const int* __restrict__ src, const int* __restrict__ dst,
                           unsigned* __restrict__ menc, int ne, int nh) {
    int i = blockIdx.x * blockDim.x + threadIdx.x;
    if (i >= ne * nh) return;
    int e = i / nh, hd = i % nh;
    int s = src[e], d = dst[e];
    float v = lrelu(el[s * nh + hd] + er[d * nh + hd], SLOPE_ATTN);
    atomicMax(&menc[d * nh + hd], enc_f(v));
}

// ex = exp(e - m[dst]); segment_sum into s
__global__ void edge_exp_k(const float* __restrict__ el, const float* __restrict__ er,
                           const int* __restrict__ src, const int* __restrict__ dst,
                           const unsigned* __restrict__ menc, float* __restrict__ ex,
                           float* __restrict__ ssum, int ne, int nh) {
    int i = blockIdx.x * blockDim.x + threadIdx.x;
    if (i >= ne * nh) return;
    int e = i / nh, hd = i % nh;
    int s = src[e], d = dst[e];
    float v = lrelu(el[s * nh + hd] + er[d * nh + hd], SLOPE_ATTN);
    float m = dec_f(menc[d * nh + hd]);
    float x = __expf(v - m);
    ex[i] = x;
    atomicAdd(&ssum[d * nh + hd], x);
}

// alpha[e,h] = ex[e,h] / ssum[dst[e],h]   (in place over ex)
__global__ void edge_alpha_k(float* __restrict__ ex, const float* __restrict__ ssum,
                             const int* __restrict__ dst, int ne, int nh) {
    int i = blockIdx.x * blockDim.x + threadIdx.x;
    if (i >= ne * nh) return;
    int e = i / nh, hd = i % nh;
    ex[i] = ex[i] / ssum[dst[e] * nh + hd];
}

// out[dst,h,d] += alpha[e,h] * hfeat[src,h,d]  -- the hot scatter kernel
__global__ void edge_agg_k(const float* __restrict__ alpha,
                           const int* __restrict__ src, const int* __restrict__ dst,
                           const float* __restrict__ hfeat, float* __restrict__ out,
                           int ne, int nh, int D) {
    int e  = blockIdx.x;
    int hd = blockIdx.y * blockDim.x + threadIdx.x;
    int HD = nh * D;
    if (e >= ne || hd >= HD) return;
    int hh = hd / D;
    int s = src[e], d = dst[e];
    float a = alpha[(size_t)e * nh + hh];
    atomicAdd(&out[(size_t)d * HD + hd], a * hfeat[(size_t)s * HD + hd]);
}

__global__ void bias_act_k(float* __restrict__ x, const float* __restrict__ bias,
                           long total, int HD, float slope) {
    long i = (long)blockIdx.x * blockDim.x + threadIdx.x;
    if (i >= total) return;
    int j = (int)(i % HD);
    x[i] = lrelu(x[i] + bias[j], slope);
}

__global__ void mean_heads_k(const float* __restrict__ in, float* __restrict__ out,
                             long total, int nh, int D) {
    long i = (long)blockIdx.x * blockDim.x + threadIdx.x;
    if (i >= total) return;
    long n = i / D;
    int  d = (int)(i % D);
    float s = 0.f;
    for (int hd = 0; hd < nh; ++hd) s += in[(n * nh + hd) * (size_t)D + d];
    out[i] = s * (1.0f / nh);
}

__global__ void fill_zero_k(float* __restrict__ p, long n) {
    long i = (long)blockIdx.x * blockDim.x + threadIdx.x;
    if (i < n) p[i] = 0.f;
}

// ---------------------------------------------------------------------------
// Host-side orchestration
// ---------------------------------------------------------------------------
static inline int cdiv(long a, long b) { return (int)((a + b - 1) / b); }

template <int NnC>
static void launch_gemm_n(const float* A, const float* B, float* C,
                          int M, int K, hipStream_t st) {
    long waves = (long)((M + 15) / 16) * ((NnC + 31) / 32);
    gemm_wmma<NnC, false, false><<<cdiv(waves, 4), 128, 0, st>>>(A, B, C, M, K);
}

static void launch_gat_edges(const float* hlin, const float* al, const float* ar,
                             const int* src, const int* dst, int ne,
                             float* el, float* er, unsigned* menc, float* ssum,
                             float* ex, float* agg, int D, hipStream_t st) {
    attn_dots<<<cdiv((long)NN * HH, 4), 128, 0, st>>>(hlin, al, ar, el, er, NN, HH, D);
    // zero accumulators (menc as float-zero == uint 0 == below any real encoding)
    fill_zero_k<<<cdiv((long)NN * HH, 256), 256, 0, st>>>((float*)menc, (long)NN * HH);
    fill_zero_k<<<cdiv((long)NN * HH, 256), 256, 0, st>>>(ssum, (long)NN * HH);
    fill_zero_k<<<cdiv((long)NN * HH * D, 256), 256, 0, st>>>(agg, (long)NN * HH * D);
    // edge softmax + aggregation
    edge_max_k<<<cdiv((long)ne * HH, 256), 256, 0, st>>>(el, er, src, dst, menc, ne, HH);
    edge_exp_k<<<cdiv((long)ne * HH, 256), 256, 0, st>>>(el, er, src, dst, menc, ex, ssum, ne, HH);
    edge_alpha_k<<<cdiv((long)ne * HH, 256), 256, 0, st>>>(ex, ssum, dst, ne, HH);
    dim3 gagg(ne, cdiv((long)HH * D, 256));
    edge_agg_k<<<gagg, 256, 0, st>>>(ex, src, dst, hlin, agg, ne, HH, D);
}

extern "C" void kernel_launch(void* const* d_in, const int* in_sizes, int n_in,
                              void* d_out, int out_size, void* d_ws, size_t ws_size,
                              hipStream_t stream) {
    const float* feat = (const float*)d_in[0];
    const int*   src  = (const int*)d_in[1];
    const int*   dst  = (const int*)d_in[2];
    const float* W1   = (const float*)d_in[3];
    const float* al1  = (const float*)d_in[4];
    const float* ar1  = (const float*)d_in[5];
    const float* b1   = (const float*)d_in[6];
    const float* W2   = (const float*)d_in[7];
    const float* al2  = (const float*)d_in[8];
    const float* ar2  = (const float*)d_in[9];
    const float* b2   = (const float*)d_in[10];
    const float* W3   = (const float*)d_in[11];
    const float* al3  = (const float*)d_in[12];
    const float* ar3  = (const float*)d_in[13];
    const float* b3   = (const float*)d_in[14];
    const int ne = in_sizes[1];            // E + N (self loops appended)

    // ---- workspace layout (floats) ----
    float* ws = (float*)d_ws;
    size_t off = 0;
    auto alloc = [&](size_t n) { float* p = ws + off; off += n; return p; };
    float*    h1   = alloc((size_t)NN * HH * HID);
    float*    el1v = alloc((size_t)NN * HH);
    float*    er1v = alloc((size_t)NN * HH);
    unsigned* m1   = (unsigned*)alloc((size_t)NN * HH);
    float*    s1   = alloc((size_t)NN * HH);
    float*    ex1  = alloc((size_t)ne * HH);
    float*    agg1 = alloc((size_t)NN * HH * HID);
    float*    h2   = alloc((size_t)NN * HH * OUTD);
    float*    el2v = alloc((size_t)NN * HH);
    float*    er2v = alloc((size_t)NN * HH);
    unsigned* m2   = (unsigned*)alloc((size_t)NN * HH);
    float*    s2   = alloc((size_t)NN * HH);
    float*    ex2  = alloc((size_t)ne * HH);
    float*    agg2 = alloc((size_t)NN * HH * OUTD);
    float*    hme  = alloc((size_t)NN * OUTD);
    float*    h3   = alloc((size_t)NN * HH * FF);
    float*    el3v = alloc((size_t)NN * HH);
    float*    er3v = alloc((size_t)NN * HH);
    unsigned* m3   = (unsigned*)alloc((size_t)NN * HH);
    float*    s3   = alloc((size_t)NN * HH);
    float*    ex3  = alloc((size_t)ne * HH);
    float*    agg3 = alloc((size_t)NN * HH * FF);

    float* ahat = (float*)d_out;                   // [N,N]
    float* xhat = ahat + (size_t)NN * NN;          // [N,F]

    // ---- encoder layer 1: F -> [N,H,HID] ----
    launch_gemm_n<HH * HID>(feat, W1, h1, NN, FF, stream);
    launch_gat_edges(h1, al1, ar1, src, dst, ne, el1v, er1v, m1, s1, ex1, agg1, HID, stream);
    bias_act_k<<<cdiv((long)NN * HH * HID, 256), 256, 0, stream>>>(agg1, b1,
        (long)NN * HH * HID, HH * HID, SLOPE_ACT);

    // ---- encoder layer 2: [N,H*HID] -> [N,H,OUT] -> mean heads -> [N,OUT] ----
    launch_gemm_n<HH * OUTD>(agg1, W2, h2, NN, HH * HID, stream);
    launch_gat_edges(h2, al2, ar2, src, dst, ne, el2v, er2v, m2, s2, ex2, agg2, OUTD, stream);
    bias_act_k<<<cdiv((long)NN * HH * OUTD, 256), 256, 0, stream>>>(agg2, b2,
        (long)NN * HH * OUTD, HH * OUTD, SLOPE_ACT);
    mean_heads_k<<<cdiv((long)NN * OUTD, 256), 256, 0, stream>>>(agg2, hme,
        (long)NN * OUTD, HH, OUTD);

    // ---- attribute decoder: [N,OUT] -> [N,H,F] -> mean heads -> x_hat ----
    launch_gemm_n<HH * FF>(hme, W3, h3, NN, OUTD, stream);
    launch_gat_edges(h3, al3, ar3, src, dst, ne, el3v, er3v, m3, s3, ex3, agg3, FF, stream);
    bias_act_k<<<cdiv((long)NN * HH * FF, 256), 256, 0, stream>>>(agg3, b3,
        (long)NN * HH * FF, HH * FF, SLOPE_ACT);
    mean_heads_k<<<cdiv((long)NN * FF, 256), 256, 0, stream>>>(agg3, xhat,
        (long)NN * FF, HH, FF);

    // ---- structure decoder: a_hat = h @ h.T (WMMA, nontemporal 268MB store) ----
    {
        long waves = (long)(NN / 16) * (NN / 32);
        gemm_wmma<NN, true, true><<<cdiv(waves, 4), 128, 0, stream>>>(hme, hme, ahat,
                                                                     NN, OUTD);
    }
}
